// QANet_53824530153751
// MI455X (gfx1250) — matrix operand order repaired and checked by reading, not compiled, not run on hardware
//
#include <hip/hip_runtime.h>
#include <hip/hip_bf16.h>
#include <math.h>

typedef __attribute__((ext_vector_type(16))) _Float16 v16h;
typedef __attribute__((ext_vector_type(8)))  _Float16 v8h;
typedef __attribute__((ext_vector_type(8)))  float    v8f;

static constexpr int kB  = 24;
static constexpr int kD  = 64;
static constexpr int kH  = 8;
static constexpr int kN  = 256;
static constexpr int kM  = 64;
static constexpr int kCN = 16;
static constexpr int kCE = 200;
static constexpr int kWE = 300;
static constexpr int kCIN1 = kWE + kD; // 364

// ---------------------------------------------------------------------------
// WMMA fragment loaders (ISA 7.12.2 layouts, wave32) — vectorized LDS reads
// ---------------------------------------------------------------------------
static __device__ __forceinline__ v16h frag_a(const _Float16* lA, int lda, int k0, int lane) {
  int m = lane & 15, half = lane >> 4;
  const _Float16* row = lA + m * lda + k0 + half * 8;
  v8h lo = *(const v8h*)(row);        // K = k0 + half*8 + [0..7]
  v8h hi = *(const v8h*)(row + 16);   // K = k0 + 16 + half*8 + [0..7]
  return __builtin_shufflevector(lo, hi, 0, 1, 2, 3, 4, 5, 6, 7, 8, 9, 10, 11, 12, 13, 14, 15);
}
// B staged in fragment order: [nt][kq][lane][16 halves] -> two b128 loads
static __device__ __forceinline__ v16h frag_bf(const _Float16* lBf, int nt_l, int kq, int lane) {
  const _Float16* p = lBf + ((((nt_l << 1) + kq) << 5) + lane) * 16;
  v8h lo = *(const v8h*)p;
  v8h hi = *(const v8h*)(p + 8);
  return __builtin_shufflevector(lo, hi, 0, 1, 2, 3, 4, 5, 6, 7, 8, 9, 10, 11, 12, 13, 14, 15);
}

// ---------------------------------------------------------------------------
// Generic batched GEMM: C[b2,b1] = op(A[b2,b1] x B[b2,b1]) (+bias, relu, +res)
// Arbitrary element strides; K staged in 64-wide chunks as f16 in LDS.
// 256 threads = 8 waves; each wave owns one 16x16 tile of a 64x32 macro-tile.
// ---------------------------------------------------------------------------
__global__ void __launch_bounds__(256) k_gemm(
    const float* __restrict__ A, const float* __restrict__ Bm, float* __restrict__ C,
    int Mr, int Nr, int Kr, int ntg,
    long long a_b2, long long a_b1, int a_rs, int a_cs,
    long long b_b2, long long b_b1, int b_rs, int b_cs,
    long long c_b2, long long c_b1, int c_rs, int c_cs,
    const float* __restrict__ bias, long long bias_b2, long long bias_b1,
    const float* __restrict__ R, long long r_b2, long long r_b1, int r_rs, int r_cs,
    int relu)
{
  __shared__ _Float16 lA[64 * 72];             // [row][K] row stride 72 (144B, 16B aligned)
  __shared__ _Float16 lBf[2 * 2 * 32 * 16];    // fragment-ordered B
  const int b1 = blockIdx.y, b2 = blockIdx.z;
  const int mt = blockIdx.x / ntg, nt = blockIdx.x % ntg;
  const float* Ab = A + b2 * a_b2 + b1 * a_b1;
  const float* Bb = Bm + b2 * b_b2 + b1 * b_b1;
  float* Cb = C + b2 * c_b2 + b1 * c_b1;
  const int tid = threadIdx.x, lane = tid & 31, wave = tid >> 5;
  const int mt_l = wave >> 1, nt_l = wave & 1;
  const int row0 = mt * 64, col0 = nt * 32;
  v8f acc = {};
  for (int kc = 0; kc < Kr; kc += 64) {
    // A chunk 64x64, packed f16 pairs -> 32-bit LDS stores
    for (int i = tid; i < 64 * 32; i += 256) {
      int r = i >> 5, kp = (i & 31) * 2;
      int gr = row0 + r, gk = kc + kp;
      float v0 = (gr < Mr && gk < Kr) ? Ab[(long long)gr * a_rs + (long long)gk * a_cs] : 0.f;
      float v1 = (gr < Mr && gk + 1 < Kr) ? Ab[(long long)gr * a_rs + (long long)(gk + 1) * a_cs] : 0.f;
      union { unsigned u; _Float16 h[2]; } pk;
      pk.h[0] = (_Float16)v0; pk.h[1] = (_Float16)v1;
      *(unsigned*)&lA[r * 72 + kp] = pk.u;
    }
    // B chunk 64x32 into fragment order
    for (int i = tid; i < 32 * 32; i += 256) {
      int kp = (i >> 5) * 2, n = i & 31;
      int gk = kc + kp, gn = col0 + n;
      float v0 = (gk < Kr && gn < Nr) ? Bb[(long long)gk * b_rs + (long long)gn * b_cs] : 0.f;
      float v1 = (gk + 1 < Kr && gn < Nr) ? Bb[(long long)(gk + 1) * b_rs + (long long)gn * b_cs] : 0.f;
      int nt_i = n >> 4, kq = kp >> 5, halfk = (kp >> 3) & 1, e = (kp & 7) | ((kp >> 1) & 8);
      int lanei = halfk * 16 + (n & 15);
      union { unsigned u; _Float16 h[2]; } pk;
      pk.h[0] = (_Float16)v0; pk.h[1] = (_Float16)v1;
      *(unsigned*)&lBf[(((nt_i * 2 + kq) * 32 + lanei) << 4) + e] = pk.u;
    }
    __syncthreads();
#pragma unroll
    for (int kq = 0; kq < 2; ++kq) {
      v16h a = frag_a(lA + mt_l * 16 * 72, 72, kq * 32, lane);
      v16h b = frag_bf(lBf, nt_l, kq, lane);
      acc = __builtin_amdgcn_wmma_f32_16x16x32_f16(false, a, false, b, (short)0, acc, false, false);
    }
    __syncthreads();
  }
  const int half = lane >> 4, nloc = lane & 15;
#pragma unroll
  for (int v = 0; v < 8; ++v) {
    int gr = row0 + mt_l * 16 + v + 8 * half;
    int gc = col0 + nt_l * 16 + nloc;
    if (gr < Mr && gc < Nr) {
      float f = acc[v];
      if (bias) f += bias[b2 * bias_b2 + b1 * bias_b1 + gr];
      if (relu) f = fmaxf(f, 0.f);
      if (R) f += R[b2 * r_b2 + b1 * r_b1 + (long long)gr * r_rs + (long long)gc * r_cs];
      Cb[(long long)gr * c_rs + (long long)gc * c_cs] = f;
    }
  }
}

// ---------------------------------------------------------------------------
// Char embedding: depthwise 5x5 conv over (pos,char) + pointwise CE->D + relu
// + max over char slots. One block per (b, n).
// ---------------------------------------------------------------------------
__global__ void __launch_bounds__(256) k_char_embed(
    const float* __restrict__ char_table, const int* __restrict__ cid,
    const float* __restrict__ e2dw, const float* __restrict__ e2db,
    const float* __restrict__ e2pw, const float* __restrict__ e2pb,
    float* __restrict__ out, int L)
{
  const int n = blockIdx.x, b = blockIdx.y, tid = threadIdx.x;
  __shared__ int   cid_s[5 * 16];
  __shared__ float DW[16 * kCE];
  __shared__ float red[4 * 64];
  if (tid < 80) {
    int dn = tid / 16, k = tid % 16;
    int nn = n + dn - 2;
    cid_s[tid] = (nn >= 0 && nn < L) ? cid[((long long)b * L + nn) * kCN + k] : -1;
  }
  __syncthreads();
  for (int i = tid; i < 16 * kCE; i += 256) {
    int k = i / kCE, c = i % kCE;
    float s = e2db[c];
    for (int dn = 0; dn < 5; ++dn)
      for (int dk = 0; dk < 5; ++dk) {
        int kk = k + dk - 2;
        if (kk >= 0 && kk < 16) {
          int cc = cid_s[dn * 16 + kk];
          if (cc >= 0) s += char_table[cc * kCE + c] * e2dw[c * 25 + dn * 5 + dk];
        }
      }
    DW[i] = s;
  }
  __syncthreads();
  const int d = tid & 63, kg = tid >> 6;
  float mx = 0.f; // relu values are >= 0
  for (int k = kg * 4; k < kg * 4 + 4; ++k) {
    float s = e2pb[d];
    for (int c = 0; c < kCE; ++c) s += e2pw[d * kCE + c] * DW[k * kCE + c];
    mx = fmaxf(mx, fmaxf(s, 0.f));
  }
  red[kg * 64 + d] = mx;
  __syncthreads();
  if (tid < 64) {
    float m2 = fmaxf(fmaxf(red[tid], red[64 + tid]), fmaxf(red[128 + tid], red[192 + tid]));
    out[((long long)b * kD + tid) * L + n] = m2;
  }
}

// emb[b, c, l] : c<64 -> char features ; else word_table[wid][c-64]
__global__ void k_build_emb(const float* __restrict__ charb, const float* __restrict__ word_table,
                            const int* __restrict__ wid, float* __restrict__ emb, int L, long long tot)
{
  long long idx = blockIdx.x * 256LL + threadIdx.x;
  if (idx >= tot) return;
  int l = (int)(idx % L);
  int c = (int)((idx / L) % kCIN1);
  int b = (int)(idx / ((long long)L * kCIN1));
  float v;
  if (c < kD) v = charb[((long long)b * kD + c) * L + l];
  else        v = word_table[(long long)wid[(long long)b * L + l] * kWE + (c - kD)];
  emb[idx] = v;
}

// depthwise 1-D conv, same padding
__global__ void k_dwconv1d(const float* __restrict__ in, const float* __restrict__ w,
                           const float* __restrict__ bias, float* __restrict__ out,
                           int Cc, int L, int ksz, long long tot)
{
  long long idx = blockIdx.x * 256LL + threadIdx.x;
  if (idx >= tot) return;
  int l = (int)(idx % L);
  int c = (int)((idx / L) % Cc);
  long long bc = idx / L;
  int pad = ksz / 2;
  float s = bias ? bias[c] : 0.f;
  for (int t = 0; t < ksz; ++t) {
    int ll = l + t - pad;
    if (ll >= 0 && ll < L) s += in[bc * L + ll] * w[c * ksz + t];
  }
  out[idx] = s;
}

// row-norm over the length dimension (mean, var with (L-1) divisor)
__global__ void __launch_bounds__(256) k_norm(const float* __restrict__ in, float* __restrict__ out, int L)
{
  const int row = blockIdx.x, tid = threadIdx.x;
  __shared__ float red[256];
  const float* x = in + (long long)row * L;
  float v = (tid < L) ? x[tid] : 0.f;
  red[tid] = v; __syncthreads();
  for (int s = 128; s > 0; s >>= 1) { if (tid < s) red[tid] += red[tid + s]; __syncthreads(); }
  float mu = red[0] / L;
  __syncthreads();
  float d = (tid < L) ? (x[tid] - mu) : 0.f;
  red[tid] = d * d; __syncthreads();
  for (int s = 128; s > 0; s >>= 1) { if (tid < s) red[tid] += red[tid + s]; __syncthreads(); }
  float var = red[0] / (L - 1);
  float inv = 1.f / (sqrtf(var) + 1e-6f);
  if (tid < L) out[(long long)row * L + tid] = d * inv;
}

// x += sin(sin(pos * freq(d) + phase(d)))
__global__ void k_posenc(float* __restrict__ x, int L, long long tot)
{
  long long idx = blockIdx.x * 256LL + threadIdx.x;
  if (idx >= tot) return;
  int l = (int)(idx % L);
  int d = (int)((idx / L) % kD);
  int de = d & ~1;
  float f = powf(10000.f, -(float)de / (float)kD);
  if (d & 1) f = -f;
  float ph = (d & 1) ? 1.5707963267948966f : 0.f;
  x[idx] += sinf(sinf((float)l * f + ph));
}

// fused attention core: per (h,b) column-wise softmax over contracted index l.
// Q/K/V tiles (8 x L each) are brought into LDS with CDNA5 async-to-LDS DMA
// (ASYNCcnt-tracked), then each thread owns one output column m.
__global__ void __launch_bounds__(256) k_att_core(
    const float* __restrict__ Q, const float* __restrict__ K, const float* __restrict__ V,
    float* __restrict__ head, int L)
{
  extern __shared__ float smem[];
  float* Qs = smem;
  float* Ks = smem + 8 * L;
  float* Vs = smem + 16 * L;
  const long long base = (long long)blockIdx.x * 8 * L;
  const int tid = threadIdx.x;
  // async copy: each active lane moves 16B per issue; LDS offsets are relative
  // to the dynamic-LDS base (static LDS is zero in this kernel).
  for (int i = tid * 4; i < 8 * L; i += 256 * 4) {
    unsigned qoff = (unsigned)(i * 4);
    unsigned koff = (unsigned)((8 * L + i) * 4);
    unsigned voff = (unsigned)((16 * L + i) * 4);
    unsigned long long qa = (unsigned long long)(const void*)(Q + base + i);
    unsigned long long ka = (unsigned long long)(const void*)(K + base + i);
    unsigned long long va = (unsigned long long)(const void*)(V + base + i);
    asm volatile("global_load_async_to_lds_b128 %0, %1, off" :: "v"(qoff), "v"(qa) : "memory");
    asm volatile("global_load_async_to_lds_b128 %0, %1, off" :: "v"(koff), "v"(ka) : "memory");
    asm volatile("global_load_async_to_lds_b128 %0, %1, off" :: "v"(voff), "v"(va) : "memory");
  }
  asm volatile("s_wait_asynccnt 0x0" ::: "memory");
  __syncthreads();
  const float inv = 0.3535533905932738f; // 1/sqrt(8)
  const int h = blockIdx.x / kB, b = blockIdx.x % kB;
  for (int m = tid; m < L; m += 256) {
    float kk[8];
#pragma unroll
    for (int v = 0; v < 8; ++v) kk[v] = Ks[v * L + m];
    float mx = -1e30f;
    for (int l = 0; l < L; ++l) {
      float s = 0.f;
#pragma unroll
      for (int v = 0; v < 8; ++v) s += Qs[v * L + l] * kk[v];
      mx = fmaxf(mx, s * inv);
    }
    float den = 0.f, num[8];
#pragma unroll
    for (int v = 0; v < 8; ++v) num[v] = 0.f;
    for (int l = 0; l < L; ++l) {
      float s = 0.f;
#pragma unroll
      for (int v = 0; v < 8; ++v) s += Qs[v * L + l] * kk[v];
      float e = expf(s * inv - mx);
      den += e;
#pragma unroll
      for (int v = 0; v < 8; ++v) num[v] += e * Vs[v * L + l];
    }
    float id = 1.f / den;
#pragma unroll
    for (int v = 0; v < 8; ++v)
      head[((long long)b * kD + h * 8 + v) * L + m] = num[v] * id;
  }
}

// highway combine: x = sigmoid(G)*NL + (1-sigmoid(G))*x  (NL already relu'd)
__global__ void k_hwy(float* __restrict__ x, const float* __restrict__ G,
                      const float* __restrict__ NL, long long tot)
{
  long long idx = blockIdx.x * 256LL + threadIdx.x;
  if (idx >= tot) return;
  float g = 1.f / (1.f + expf(-G[idx]));
  x[idx] = g * NL[idx] + (1.f - g) * x[idx];
}

// out[b,l] = sum_d w[b*wstride + woff + d] * X[b,d,l]
__global__ void k_chan_dot(const float* __restrict__ X, const float* __restrict__ w,
                           int wstride, int woff, float* __restrict__ out, int L, long long tot)
{
  long long idx = blockIdx.x * 256LL + threadIdx.x;
  if (idx >= tot) return;
  int l = (int)(idx % L);
  int b = (int)(idx / L);
  float s = 0.f;
  for (int d = 0; d < kD; ++d)
    s += w[b * wstride + woff + d] * X[((long long)b * kD + d) * L + l];
  out[idx] = s;
}

// out[b,d,l] = X[b,d,l] * w[b*wstride + woff + d]
__global__ void k_scale_chan(const float* __restrict__ X, const float* __restrict__ w,
                             int wstride, int woff, float* __restrict__ out, int L, long long tot)
{
  long long idx = blockIdx.x * 256LL + threadIdx.x;
  if (idx >= tot) return;
  int d = (int)((idx / L) % kD);
  int b = (int)(idx / ((long long)L * kD));
  out[idx] = X[idx] * w[b * wstride + woff + d];
}

// S[b,n,m] += cvec[b,n] + qvec[b,m]
__global__ void k_add_rank1(float* __restrict__ S, const float* __restrict__ cvec,
                            const float* __restrict__ qvec, long long tot)
{
  long long idx = blockIdx.x * 256LL + threadIdx.x;
  if (idx >= tot) return;
  int m = (int)(idx % kM);
  int n = (int)((idx / kM) % kN);
  int b = (int)(idx / ((long long)kM * kN));
  S[idx] += cvec[b * kN + n] + qvec[b * kM + m];
}

// generic strided softmax: row r -> base = (r/inner)*outer + (r%inner)*inner_stride
__global__ void k_softmax_gen(const float* __restrict__ in, float* __restrict__ out,
                              int count, int len, int inner,
                              long long outer_stride, long long inner_stride, long long elem_stride)
{
  int r = blockIdx.x * 256 + threadIdx.x;
  if (r >= count) return;
  long long base = (long long)(r / inner) * outer_stride + (long long)(r % inner) * inner_stride;
  float mx = -1e30f;
  for (int j = 0; j < len; ++j) mx = fmaxf(mx, in[base + j * elem_stride]);
  float den = 0.f;
  for (int j = 0; j < len; ++j) den += expf(in[base + j * elem_stride] - mx);
  float id = 1.f / den;
  for (int j = 0; j < len; ++j) out[base + j * elem_stride] = expf(in[base + j * elem_stride] - mx) * id;
}

// X4 = concat([C, A, C*A, C*Bm], channel axis), all (B,64,256)
__global__ void k_x4(const float* __restrict__ C, const float* __restrict__ A,
                     const float* __restrict__ Bm, float* __restrict__ X4, long long tot)
{
  long long idx = blockIdx.x * 256LL + threadIdx.x;
  if (idx >= tot) return;
  int n = (int)(idx % kN);
  int r = (int)((idx / kN) % 256);
  int b = (int)(idx / ((long long)kN * 256));
  int quad = r >> 6, d = r & 63;
  long long o = ((long long)b * kD + d) * kN + n;
  float c = C[o], v;
  if (quad == 0) v = c;
  else if (quad == 1) v = A[o];
  else if (quad == 2) v = c * A[o];
  else v = c * Bm[o];
  X4[idx] = v;
}

// Y[b,n] = sum_d w[b,d]*M0 + w[b,64+d]*Mx
__global__ void k_pdot(const float* __restrict__ M0, const float* __restrict__ Mx,
                       const float* __restrict__ w, float* __restrict__ Y, long long tot)
{
  long long idx = blockIdx.x * 256LL + threadIdx.x;
  if (idx >= tot) return;
  int n = (int)(idx % kN);
  int b = (int)(idx / kN);
  float s = 0.f;
  for (int d = 0; d < kD; ++d) {
    long long o = ((long long)b * kD + d) * kN + n;
    s += w[b * 128 + d] * M0[o] + w[b * 128 + 64 + d] * Mx[o];
  }
  Y[idx] = s;
}

// ---------------------------------------------------------------------------
// Host orchestration
// ---------------------------------------------------------------------------
static inline dim3 gs(long long n) { return dim3((unsigned)((n + 255) / 256)); }

static inline void launch_gemm(hipStream_t st,
    const float* A, long long a_b2, long long a_b1, int a_rs, int a_cs,
    const float* B, long long b_b2, long long b_b1, int b_rs, int b_cs,
    float* C, long long c_b2, long long c_b1, int c_rs, int c_cs,
    int Mr, int Nr, int Kr, int nb2, int nb1,
    const float* bias, long long bias_b2, long long bias_b1,
    const float* R, long long r_b2, long long r_b1, int r_rs, int r_cs,
    int relu)
{
  int mt = (Mr + 63) / 64, ntg = (Nr + 31) / 32;
  dim3 grid(mt * ntg, nb1, nb2);
  k_gemm<<<grid, 256, 0, st>>>(A, B, C, Mr, Nr, Kr, ntg,
      a_b2, a_b1, a_rs, a_cs, b_b2, b_b1, b_rs, b_cs, c_b2, c_b1, c_rs, c_cs,
      bias, bias_b2, bias_b1, R, r_b2, r_b1, r_rs, r_cs, relu);
}

struct EncW {
  const float* dw; const float* pw;
  const float* wq; const float* wk; const float* wv;
  const float* wo; const float* w;
  int nconv; int ksz;
};

extern "C" void kernel_launch(void* const* d_in, const int* in_sizes, int n_in,
                              void* d_out, int out_size, void* d_ws, size_t ws_size,
                              hipStream_t stream) {
  (void)in_sizes; (void)n_in; (void)out_size; (void)ws_size;
  const float* word_table = (const float*)d_in[0];
  const float* char_table = (const float*)d_in[1];
  const float* e2dw = (const float*)d_in[2];
  const float* e2db = (const float*)d_in[3];
  const float* e2pw = (const float*)d_in[4];
  const float* e2pb = (const float*)d_in[5];
  const float* e1dw = (const float*)d_in[6];
  const float* e1db = (const float*)d_in[7];
  const float* e1pw = (const float*)d_in[8];
  const float* e1pb = (const float*)d_in[9];
  const float* hlw  = (const float*)d_in[10];
  const float* hlb  = (const float*)d_in[11];
  const float* hgw  = (const float*)d_in[12];
  const float* hgb  = (const float*)d_in[13];
  EncW encw[5];
  for (int p = 0; p < 5; ++p) {
    int base = 14 + p * 7;
    encw[p].dw = (const float*)d_in[base + 0];
    encw[p].pw = (const float*)d_in[base + 1];
    encw[p].wq = (const float*)d_in[base + 2];
    encw[p].wk = (const float*)d_in[base + 3];
    encw[p].wv = (const float*)d_in[base + 4];
    encw[p].wo = (const float*)d_in[base + 5];
    encw[p].w  = (const float*)d_in[base + 6];
    encw[p].nconv = (p < 2) ? 4 : 2;
    encw[p].ksz   = (p < 2) ? 7 : 5;
  }
  const float* cq_w  = (const float*)d_in[49];
  const float* rs_dw = (const float*)d_in[50];
  const float* rs_pw = (const float*)d_in[51];
  const float* p_w0  = (const float*)d_in[52];
  const float* p_w1  = (const float*)d_in[53];
  const int* Cwid = (const int*)d_in[54];
  const int* Ccid = (const int*)d_in[55];
  const int* Qwid = (const int*)d_in[56];
  const int* Qcid = (const int*)d_in[57];
  float* outp = (float*)d_out;

  // -------- workspace bump allocator --------
  char* wsp = (char*)d_ws;
  auto alloc = [&](long long elems) -> float* {
    float* p = (float*)wsp;
    wsp += ((elems * 4 + 255) / 256) * 256;
    return p;
  };
  float* charbC = alloc((long long)kB * kD * kN);
  float* charbQ = alloc((long long)kB * kD * kM);
  float* embT   = alloc((long long)kB * kCIN1 * kN);
  float* dwT    = alloc((long long)kB * kCIN1 * kN);
  float* CA     = alloc((long long)kB * kD * kN);
  float* CBuf   = alloc((long long)kB * kD * kN);
  float* QA     = alloc((long long)kB * kD * kM);
  float* QB     = alloc((long long)kB * kD * kM);
  float* Tn     = alloc((long long)kB * kD * kN);
  float* Td     = alloc((long long)kB * kD * kN);
  float* Tg     = alloc((long long)kB * kD * kN);
  float* Qb     = alloc((long long)kH * kB * 8 * kN);
  float* Kb     = alloc((long long)kH * kB * 8 * kN);
  float* Vb     = alloc((long long)kH * kB * 8 * kN);
  float* headb  = alloc((long long)kB * kD * kN);
  float* cvec   = alloc((long long)kB * kN);
  float* qvec   = alloc((long long)kB * kM);
  float* Cw     = alloc((long long)kB * kD * kN);
  float* S      = alloc((long long)kB * kN * kM);
  float* S1     = alloc((long long)kB * kN * kM);
  float* S2     = alloc((long long)kB * kN * kM);
  float* Aatt   = alloc((long long)kB * kD * kN);
  float* Bmb    = alloc((long long)kB * kD * kN);
  float* X      = alloc((long long)kB * 256 * kN);
  float* X4     = alloc((long long)kB * 256 * kN);
  float* M0b    = alloc((long long)kB * kD * kN);
  float* M1b    = alloc((long long)kB * kD * kN);
  float* MA     = alloc((long long)kB * kD * kN);
  float* MB     = alloc((long long)kB * kD * kN);
  float* Y0     = alloc((long long)kB * kN);
  float* Y1     = alloc((long long)kB * kN);

  // -------- embedding pipeline (char conv -> concat -> dwsep1d -> highway) --------
  auto embed_pipeline = [&](const int* wid, const int* cid, int L, float* charb, float* st) {
    dim3 g(L, kB);
    k_char_embed<<<g, 256, 0, stream>>>(char_table, cid, e2dw, e2db, e2pw, e2pb, charb, L);
    long long tot364 = (long long)kB * kCIN1 * L;
    k_build_emb<<<gs(tot364), 256, 0, stream>>>(charb, word_table, wid, embT, L, tot364);
    k_dwconv1d<<<gs(tot364), 256, 0, stream>>>(embT, e1dw, e1db, dwT, kCIN1, L, 5, tot364);
    launch_gemm(stream, e1pw, 0, 0, kCIN1, 1,
                dwT, 0, (long long)kCIN1 * L, L, 1,
                st, 0, (long long)kD * L, L, 1,
                kD, L, kCIN1, 1, kB,
                e1pb, 0, 0, nullptr, 0, 0, 0, 0, 0);
    long long tot64 = (long long)kB * kD * L;
    for (int i = 0; i < 2; ++i) {
      launch_gemm(stream, hgw + i * kD * kD, 0, 0, kD, 1,
                  st, 0, (long long)kD * L, L, 1,
                  Tg, 0, (long long)kD * L, L, 1,
                  kD, L, kD, 1, kB,
                  hgb + i * kD, 0, 0, nullptr, 0, 0, 0, 0, 0);
      launch_gemm(stream, hlw + i * kD * kD, 0, 0, kD, 1,
                  st, 0, (long long)kD * L, L, 1,
                  Td, 0, (long long)kD * L, L, 1,
                  kD, L, kD, 1, kB,
                  hlb + i * kD, 0, 0, nullptr, 0, 0, 0, 0, 1);
      k_hwy<<<gs(tot64), 256, 0, stream>>>(st, Tg, Td, tot64);
    }
  };

  // -------- encoder block --------
  auto enc_block = [&](float** pcur, float** palt, const EncW& W, int L) {
    float* cur = *pcur; float* alt = *palt;
    long long tot64 = (long long)kB * kD * L;
    k_posenc<<<gs(tot64), 256, 0, stream>>>(cur, L, tot64);
    for (int i = 0; i < W.nconv; ++i) {
      k_norm<<<dim3(kB * kD), 256, 0, stream>>>(cur, Tn, L);
      k_dwconv1d<<<gs(tot64), 256, 0, stream>>>(Tn, W.dw + (long long)i * kD * W.ksz, nullptr, Td, kD, L, W.ksz, tot64);
      launch_gemm(stream, W.pw + (long long)i * kD * kD, 0, 0, kD, 1,
                  Td, 0, (long long)kD * L, L, 1,
                  alt, 0, (long long)kD * L, L, 1,
                  kD, L, kD, 1, kB,
                  nullptr, 0, 0,
                  cur, 0, (long long)kD * L, L, 1, 1);
      float* t = cur; cur = alt; alt = t;
    }
    // self-attention
    k_norm<<<dim3(kB * kD), 256, 0, stream>>>(cur, Tn, L);
    const float* pw3[3] = { W.wq, W.wk, W.wv };
    float* dst3[3] = { Qb, Kb, Vb };
    for (int j = 0; j < 3; ++j)
      launch_gemm(stream, pw3[j], (long long)kB * 8 * kD, 8 * kD, kD, 1,
                  Tn, 0, (long long)kD * L, L, 1,
                  dst3[j], (long long)kB * 8 * L, (long long)8 * L, L, 1,
                  8, L, kD, kH, kB,
                  nullptr, 0, 0, nullptr, 0, 0, 0, 0, 0);
    k_att_core<<<dim3(kH * kB), 256, (size_t)(24 * L * sizeof(float)), stream>>>(Qb, Kb, Vb, headb, L);
    launch_gemm(stream, W.wo, 0, (long long)kD * kD, kD, 1,
                headb, 0, (long long)kD * L, L, 1,
                alt, 0, (long long)kD * L, L, 1,
                kD, L, kD, 1, kB,
                nullptr, 0, 0,
                cur, 0, (long long)kD * L, L, 1, 0);
    { float* t = cur; cur = alt; alt = t; }
    // feed-forward
    k_norm<<<dim3(kB * kD), 256, 0, stream>>>(cur, Tn, L);
    launch_gemm(stream, W.w, 0, (long long)kD * kD, kD, 1,
                Tn, 0, (long long)kD * L, L, 1,
                alt, 0, (long long)kD * L, L, 1,
                kD, L, kD, 1, kB,
                nullptr, 0, 0,
                cur, 0, (long long)kD * L, L, 1, 1);
    { float* t = cur; cur = alt; alt = t; }
    *pcur = cur; *palt = alt;
  };

  // -------- C and Q embedding + embedding encoders --------
  float* curC = CA; float* altC = CBuf;
  embed_pipeline(Cwid, Ccid, kN, charbC, curC);
  enc_block(&curC, &altC, encw[0], kN);

  float* curQ = QA; float* altQ = QB;
  embed_pipeline(Qwid, Qcid, kM, charbQ, curQ);
  enc_block(&curQ, &altQ, encw[1], kM);

  // -------- context-query attention --------
  k_chan_dot<<<gs((long long)kB * kN), 256, 0, stream>>>(curC, cq_w, 3 * kD, kD, cvec, kN, (long long)kB * kN);
  k_chan_dot<<<gs((long long)kB * kM), 256, 0, stream>>>(curQ, cq_w, 3 * kD, 0, qvec, kM, (long long)kB * kM);
  k_scale_chan<<<gs((long long)kB * kD * kN), 256, 0, stream>>>(curC, cq_w, 3 * kD, 2 * kD, Cw, kN, (long long)kB * kD * kN);
  // S[n,m] = sum_d Cw[d,n] * Q[d,m]
  launch_gemm(stream, Cw, 0, (long long)kD * kN, 1, kN,
              curQ, 0, (long long)kD * kM, kM, 1,
              S, 0, (long long)kN * kM, kM, 1,
              kN, kM, kD, 1, kB,
              nullptr, 0, 0, nullptr, 0, 0, 0, 0, 0);
  k_add_rank1<<<gs((long long)kB * kN * kM), 256, 0, stream>>>(S, cvec, qvec, (long long)kB * kN * kM);
  k_softmax_gen<<<gs(kB * kN), 256, 0, stream>>>(S, S1, kB * kN, kM, kN, (long long)kN * kM, kM, 1);
  k_softmax_gen<<<gs(kB * kM), 256, 0, stream>>>(S, S2, kB * kM, kN, kM, (long long)kN * kM, 1, kM);
  // A[d,n] = sum_m Q[d,m] * S1[n,m]
  launch_gemm(stream, curQ, 0, (long long)kD * kM, kM, 1,
              S1, 0, (long long)kN * kM, 1, kM,
              Aatt, 0, (long long)kD * kN, kN, 1,
              kD, kN, kM, 1, kB,
              nullptr, 0, 0, nullptr, 0, 0, 0, 0, 0);
  // X[n,k] = sum_m S1[n,m] * S2[k,m]
  launch_gemm(stream, S1, 0, (long long)kN * kM, kM, 1,
              S2, 0, (long long)kN * kM, 1, kM,
              X, 0, (long long)kN * kN, kN, 1,
              kN, kN, kM, 1, kB,
              nullptr, 0, 0, nullptr, 0, 0, 0, 0, 0);
  // Bm[d,k] = sum_n C[d,n] * X[n,k]
  launch_gemm(stream, curC, 0, (long long)kD * kN, kN, 1,
              X, 0, (long long)kN * kN, kN, 1,
              Bmb, 0, (long long)kD * kN, kN, 1,
              kD, kN, kN, 1, kB,
              nullptr, 0, 0, nullptr, 0, 0, 0, 0, 0);
  long long totX4 = (long long)kB * 256 * kN;
  k_x4<<<gs(totX4), 256, 0, stream>>>(curC, Aatt, Bmb, X4, totX4);

  // -------- resize dwsep conv: 256 -> 64 channels --------
  k_dwconv1d<<<gs(totX4), 256, 0, stream>>>(X4, rs_dw, nullptr, X, 256, kN, 5, totX4);
  launch_gemm(stream, rs_pw, 0, 0, 256, 1,
              X, 0, (long long)256 * kN, kN, 1,
              MA, 0, (long long)kD * kN, kN, 1,
              kD, kN, 256, 1, kB,
              nullptr, 0, 0, nullptr, 0, 0, 0, 0, 0);

  // -------- 3 x 7 model encoder blocks --------
  float* mcur = MA; float* malt = MB;
  for (int i = 0; i < 7; ++i) enc_block(&mcur, &malt, encw[2], kN);
  hipMemcpyAsync(M0b, mcur, (size_t)kB * kD * kN * sizeof(float), hipMemcpyDeviceToDevice, stream);
  for (int i = 0; i < 7; ++i) enc_block(&mcur, &malt, encw[3], kN);
  hipMemcpyAsync(M1b, mcur, (size_t)kB * kD * kN * sizeof(float), hipMemcpyDeviceToDevice, stream);
  for (int i = 0; i < 7; ++i) enc_block(&mcur, &malt, encw[4], kN);
  // mcur == M2

  // -------- pointer network --------
  long long totY = (long long)kB * kN;
  k_pdot<<<gs(totY), 256, 0, stream>>>(M0b, M1b, p_w0, Y0, totY);
  k_pdot<<<gs(totY), 256, 0, stream>>>(M0b, mcur, p_w1, Y1, totY);
  k_softmax_gen<<<gs(kB), 256, 0, stream>>>(Y0, outp, kB, kN, 1, kN, 0, 1);
  k_softmax_gen<<<gs(kB), 256, 0, stream>>>(Y1, outp + (long long)kB * kN, kB, kN, 1, kN, 0, 1);
}
